// Decoder_47467978556088
// MI455X (gfx1250) — matrix operand order repaired and checked by reading, not compiled, not run on hardware
//
#include <hip/hip_runtime.h>
#include <math.h>

// ---------------------------------------------------------------------------
// Problem constants (from reference)
// ---------------------------------------------------------------------------
#define B_    512
#define T_    64
#define J_    32
#define H_    1024
#define Z_    128
#define DIN_  173          // J + Z + TIME_DIM
#define KCAT  1216         // pad(173->176) + H, multiple of 32
#define NG    4096         // 4*H
#define POSD  48
#define NLNK  45           // NUM_LINKS*3

typedef __bf16 bf16;
typedef __bf16 v16bf __attribute__((ext_vector_type(16)));
typedef __bf16 v8bf  __attribute__((ext_vector_type(8)));
typedef float  v8f   __attribute__((ext_vector_type(8)));

__device__ __forceinline__ bf16 f2bf(float f) {
    union { float f; unsigned u; } v; v.f = f;
    unsigned r = v.u + 0x7fffu + ((v.u >> 16) & 1u);   // round-to-nearest-even
    unsigned short h = (unsigned short)(r >> 16);
    bf16 o; __builtin_memcpy(&o, &h, 2); return o;
}
__device__ __forceinline__ float sigm(float x) { return 1.0f / (1.0f + __expf(-x)); }

// ---------------------------------------------------------------------------
// WMMA GEMM:  C[M,N] = act(A[M,K](bf16) * W[N,K](bf16)^T + bias[N])
// NT (1..4) output tiles of 16x16 per wave, compile-time constant so the
// kernel has NO data-dependent branches (EXEC stays all-ones for WMMA, no
// accumulator spills). Grids are sized exactly: M%16==0, N == waves_n*NT*16,
// total waves % 8 == 0.
// A fragment (16-bit A 16x32, ISA 7.12.2): lane m=l&15,hi=l>>4 holds
//   elems 0..7  -> K = k0 + hi*8 + e ; elems 8..15 -> K = k0 + 16 + hi*8 + e
// B fragment (32x16): lane holds col n=l&15, same K mapping as A.
// C/D: lane l, vgpr r -> row m0 + r + 8*hi, col n0 + (l&15).
// ---------------------------------------------------------------------------
template<int NT>
__global__ void gemm_wmma_bf16(const bf16* __restrict__ A, int lda,
                               const bf16* __restrict__ W, int ldw,
                               const float* __restrict__ bias,
                               float* __restrict__ Cf, bf16* __restrict__ Cb,
                               int ldc, int K, int waves_n, int relu)
{
    const int w      = blockIdx.x * 8 + (threadIdx.x >> 5);
    const int lane   = threadIdx.x & 31;
    const int mtile  = w / waves_n;
    const int ntbase = (w % waves_n) * NT;
    const int m  = lane & 15;
    const int hi = lane >> 4;
    const int m0 = mtile * 16;

    v8f acc[NT];
    #pragma unroll
    for (int nt = 0; nt < NT; ++nt) acc[nt] = (v8f){0.f,0.f,0.f,0.f,0.f,0.f,0.f,0.f};

    const bf16* Arow = A + (size_t)(m0 + m) * lda + hi * 8;
    const bf16* Brow[NT];
    #pragma unroll
    for (int nt = 0; nt < NT; ++nt)
        Brow[nt] = W + (size_t)((ntbase + nt) * 16 + m) * ldw + hi * 8;

    for (int k0 = 0; k0 < K; k0 += 32) {
        const v8bf* pA = (const v8bf*)(Arow + k0);
        v8bf alo = pA[0];
        v8bf ahi = pA[2];                          // +16 elements
        v16bf af = __builtin_shufflevector(alo, ahi,
                     0,1,2,3,4,5,6,7,8,9,10,11,12,13,14,15);
        #pragma unroll
        for (int nt = 0; nt < NT; ++nt) {
            const v8bf* pB = (const v8bf*)(Brow[nt] + k0);
            v8bf b0 = pB[0], b1 = pB[2];
            v16bf bv = __builtin_shufflevector(b0, b1,
                         0,1,2,3,4,5,6,7,8,9,10,11,12,13,14,15);
            acc[nt] = __builtin_amdgcn_wmma_f32_16x16x32_bf16(
                false, af, false, bv, (short)0, acc[nt], false, false);
        }
    }

    #pragma unroll
    for (int nt = 0; nt < NT; ++nt) {
        int n = (ntbase + nt) * 16 + m;
        float bvl = bias ? bias[n] : 0.0f;         // uniform (kernel-arg) branch
        #pragma unroll
        for (int r = 0; r < 8; ++r) {
            int row = m0 + r + 8 * hi;
            float v = acc[nt][r] + bvl;
            if (relu) v = v > 0.0f ? v : 0.0f;     // uniform branch
            if (Cf) Cf[(size_t)row * ldc + n] = v;
            if (Cb) Cb[(size_t)row * ldc + n] = f2bf(v);
        }
    }
}

// ---------------------------------------------------------------------------
// Prep kernels (run once per launch)
// ---------------------------------------------------------------------------
__global__ void cvt_bf16(const float* __restrict__ s, bf16* __restrict__ d, int n) {
    int i = blockIdx.x * blockDim.x + threadIdx.x;
    if (i < n) d[i] = f2bf(s[i]);
}
// W_cat[n][k]: k<173 -> W_ih ; 173..175 pad 0 ; 176..1199 -> W_hh ; 1200..1215 pad 0
__global__ void pack_wcat(const float* __restrict__ Wih, const float* __restrict__ Whh,
                          bf16* __restrict__ out) {
    int idx = blockIdx.x * blockDim.x + threadIdx.x;
    if (idx >= NG * KCAT) return;
    int n = idx / KCAT, k = idx % KCAT;
    float v = 0.0f;
    if (k < DIN_)                  v = Wih[(size_t)n * DIN_ + k];
    else if (k >= 176 && k < 1200) v = Whh[(size_t)n * H_ + (k - 176)];
    out[idx] = f2bf(v);
}
__global__ void pack_objW2(const float* __restrict__ s, bf16* __restrict__ d) {
    int idx = blockIdx.x * blockDim.x + threadIdx.x;
    if (idx >= 16 * H_) return;
    int r = idx / H_;
    d[idx] = f2bf(r < 3 ? s[idx] : 0.0f);
}
__global__ void vec_add(const float* a, const float* b, float* o, int n) {
    int i = blockIdx.x * blockDim.x + threadIdx.x;
    if (i < n) o[i] = a[i] + b[i];
}
__global__ void pad_b2f(const float* b2, float* o) {
    int i = blockIdx.x * blockDim.x + threadIdx.x;
    if (i < 16) o[i] = (i < 3) ? b2[i] : 0.0f;
}
__global__ void init_prev(const float* ia, float* prev) {
    int i = blockIdx.x * blockDim.x + threadIdx.x;
    if (i < B_ * J_) prev[i] = ia[i % J_];
}

// ---------------------------------------------------------------------------
// Per-step pointwise kernels
// ---------------------------------------------------------------------------
// x_cat row = [pn(32) | z(128) | time(13) | pad(3) | h(1024) | pad(16)] in bf16
__global__ void assemble_xh(const float* __restrict__ prev, const float* __restrict__ z,
                            const bf16* __restrict__ h_bf,
                            const float* __restrict__ jlo, const float* __restrict__ jup,
                            bf16* __restrict__ xh, int t)
{
    int idx = blockIdx.x * blockDim.x + threadIdx.x;
    if (idx >= B_ * KCAT) return;
    int b = idx / KCAT, k = idx % KCAT;
    bf16 o;
    if (k < 32) {
        float lo = jlo[k], up = jup[k];
        float jm = 0.5f * (up + lo), jr = 0.5f * (up - lo);
        o = f2bf((prev[b * J_ + k] - jm) / jr);
    } else if (k < 160) {
        o = f2bf(z[b * Z_ + (k - 32)]);
    } else if (k < 173) {
        int j = k - 160;
        float tt = (float)t / (float)(T_ - 1);
        float v;
        if (j == 0) v = tt;
        else {
            int band = (j - 1) >> 1;
            float ang = 2.0f * 3.14159265358979f * (float)(1 << band) * tt;
            v = (j & 1) ? __sinf(ang) : __cosf(ang);
        }
        o = f2bf(v);
    } else if (k < 176) {
        o = f2bf(0.0f);
    } else if (k < 1200) {
        o = h_bf[b * H_ + (k - 176)];
    } else {
        o = f2bf(0.0f);
    }
    xh[idx] = o;
}

// torch gate order i,f,g,o across the 4*H gates
__global__ void lstm_cell(const float* __restrict__ gates, float* __restrict__ c,
                          bf16* __restrict__ h_bf)
{
    int idx = blockIdx.x * blockDim.x + threadIdx.x;
    if (idx >= B_ * H_) return;
    int b = idx / H_, j = idx % H_;
    const float* g = gates + (size_t)b * NG;
    float gi = g[j], gf = g[H_ + j], gg = g[2 * H_ + j], go = g[3 * H_ + j];
    float cn = sigm(gf) * c[idx] + sigm(gi) * tanhf(gg);
    float hn = sigm(go) * tanhf(cn);
    c[idx] = cn;
    h_bf[idx] = f2bf(hn);
}

__global__ void joint_update(float* __restrict__ prev, const float* __restrict__ rd,
                             const float* __restrict__ dscale,
                             const float* __restrict__ jlo, const float* __restrict__ jup,
                             const float* __restrict__ obj_lin,
                             float* __restrict__ jt_out, float* __restrict__ objtr, int t)
{
    int idx = blockIdx.x * blockDim.x + threadIdx.x;
    if (idx >= B_ * J_) return;
    int b = idx / J_, j = idx % J_;
    float cur = prev[idx] + rd[idx] * dscale[j];
    jt_out[((size_t)b * T_ + t) * J_ + j] = cur;                  // unclamped, as reference
    prev[idx] = fminf(fmaxf(cur, jlo[j]), jup[j]);                // clamped feeds next step
    if (j < 3) objtr[((size_t)b * T_ + t) * 3 + j] = obj_lin[b * 16 + j];
}

// FK + position normalize + log-sigma. One thread per (b,t).
__global__ void final_post(const float* __restrict__ jt, const float* __restrict__ objtr,
                           const float* __restrict__ varlin,
                           const float* __restrict__ fkW, const float* __restrict__ fkb,
                           const float* __restrict__ pm, const float* __restrict__ ps,
                           float* __restrict__ mu, float* __restrict__ ls)
{
    int idx = blockIdx.x * blockDim.x + threadIdx.x;
    if (idx >= B_ * T_) return;
    const float* jr = jt + (size_t)idx * J_;
    float jnt[J_];
    #pragma unroll
    for (int j = 0; j < J_; ++j) jnt[j] = jr[j];
    float* om = mu + (size_t)idx * POSD;
    for (int L = 0; L < NLNK; ++L) {
        float s = fkb[L];
        #pragma unroll
        for (int j = 0; j < J_; ++j) s = fmaf(fkW[L * J_ + j], jnt[j], s);
        om[L] = (s - pm[L]) / ps[L];
    }
    for (int q = 0; q < 3; ++q) {
        float v = objtr[(size_t)idx * 3 + q];
        om[NLNK + q] = (v - pm[NLNK + q]) / ps[NLNK + q];
    }
    const float* vl = varlin + (size_t)idx * POSD;
    float* ol = ls + (size_t)idx * POSD;
    for (int n = 0; n < POSD; ++n) {
        float sg = 0.05f + 0.45f * sigm(vl[n]);
        ol[n] = __logf(sg);
    }
}

// ---------------------------------------------------------------------------
// Host orchestration (graph-capture safe: only launches on `stream`)
// Input order assumed from setup_inputs() dict flattening:
//  0:z 1:W_ih 2:W_hh 3:b_ih 4:b_hh 5:zh_W 6:zh_b 7:zc_W 8:zc_b 9:obj_W1
// 10:obj_b1 11:obj_W2 12:obj_b2 13:del_W1 14:del_b1 15:del_W2 16:del_b2
// 17:del_W3 18:del_b3 19:delta_scale 20:var_W 21:var_b 22:fk_W 23:fk_b
// 24:joint_lower 25:joint_upper 26:init_angles 27:pos_mean 28:pos_std
// ---------------------------------------------------------------------------
extern "C" void kernel_launch(void* const* d_in, const int* in_sizes, int n_in,
                              void* d_out, int out_size, void* d_ws, size_t ws_size,
                              hipStream_t stream)
{
    (void)in_sizes; (void)n_in; (void)out_size; (void)ws_size;
    const float* z      = (const float*)d_in[0];
    const float* W_ih   = (const float*)d_in[1];
    const float* W_hh   = (const float*)d_in[2];
    const float* b_ih   = (const float*)d_in[3];
    const float* b_hh   = (const float*)d_in[4];
    const float* zh_W   = (const float*)d_in[5];
    const float* zh_b   = (const float*)d_in[6];
    const float* zc_W   = (const float*)d_in[7];
    const float* zc_b   = (const float*)d_in[8];
    const float* obj_W1 = (const float*)d_in[9];
    const float* obj_b1 = (const float*)d_in[10];
    const float* obj_W2 = (const float*)d_in[11];
    const float* obj_b2 = (const float*)d_in[12];
    const float* del_W1 = (const float*)d_in[13];
    const float* del_b1 = (const float*)d_in[14];
    const float* del_W2 = (const float*)d_in[15];
    const float* del_b2 = (const float*)d_in[16];
    const float* del_W3 = (const float*)d_in[17];
    const float* del_b3 = (const float*)d_in[18];
    const float* dscale = (const float*)d_in[19];
    const float* var_W  = (const float*)d_in[20];
    const float* var_b  = (const float*)d_in[21];
    const float* fk_W   = (const float*)d_in[22];
    const float* fk_b   = (const float*)d_in[23];
    const float* jlo    = (const float*)d_in[24];
    const float* jup    = (const float*)d_in[25];
    const float* init_a = (const float*)d_in[26];
    const float* pmean  = (const float*)d_in[27];
    const float* pstd   = (const float*)d_in[28];
    float* out = (float*)d_out;

    // ---- workspace carve (~40 MB) ----
    char* base = (char*)d_ws; size_t off = 0;
    auto take = [&](size_t bytes) -> char* {
        char* r = base + off; off = (off + bytes + 255) & ~(size_t)255; return r;
    };
    bf16*  wcat = (bf16*)take((size_t)NG * KCAT * 2);
    bf16*  oW1  = (bf16*)take((size_t)H_ * H_ * 2);
    bf16*  dW1  = (bf16*)take((size_t)H_ * H_ * 2);
    bf16*  dW2  = (bf16*)take((size_t)H_ * H_ * 2);
    bf16*  dW3  = (bf16*)take((size_t)J_ * H_ * 2);
    bf16*  vW   = (bf16*)take((size_t)POSD * H_ * 2);
    bf16*  oW2p = (bf16*)take((size_t)16 * H_ * 2);
    bf16*  zhWb = (bf16*)take((size_t)H_ * Z_ * 2);
    bf16*  zcWb = (bf16*)take((size_t)H_ * Z_ * 2);
    bf16*  z_bf = (bf16*)take((size_t)B_ * Z_ * 2);
    bf16*  xh   = (bf16*)take((size_t)B_ * KCAT * 2);
    bf16*  h_bf = (bf16*)take((size_t)B_ * H_ * 2);
    bf16*  t1   = (bf16*)take((size_t)B_ * H_ * 2);
    bf16*  t2   = (bf16*)take((size_t)B_ * H_ * 2);
    float* gates = (float*)take((size_t)B_ * NG * 4);
    float* cst   = (float*)take((size_t)B_ * H_ * 4);
    float* objl  = (float*)take((size_t)B_ * 16 * 4);
    float* rdel  = (float*)take((size_t)B_ * J_ * 4);
    float* vlin  = (float*)take((size_t)B_ * T_ * POSD * 4);
    float* otr   = (float*)take((size_t)B_ * T_ * 3 * 4);
    float* prev  = (float*)take((size_t)B_ * J_ * 4);
    float* gbias = (float*)take((size_t)NG * 4);
    float* ob2p  = (float*)take((size_t)16 * 4);

    #define L1D(kern, n, ...) kern<<<((n) + 255) / 256, 256, 0, stream>>>(__VA_ARGS__)

    auto gemm = [&](const bf16* A, int lda, const bf16* Wt, int ldw, const float* bias,
                    float* Cf, bf16* Cb, int ldc, int M, int N, int K, int relu) {
        int nt16 = N / 16;
        int NT = (nt16 % 4 == 0) ? 4 : nt16;         // exact: 4096/1024 -> 4, 48->3, 32->2, 16->1
        int waves_n = nt16 / NT;
        int blocks = (M / 16) * waves_n / 8;          // M=512 -> always whole blocks
        switch (NT) {
        case 1: gemm_wmma_bf16<1><<<blocks, 256, 0, stream>>>(A, lda, Wt, ldw, bias, Cf, Cb, ldc, K, waves_n, relu); break;
        case 2: gemm_wmma_bf16<2><<<blocks, 256, 0, stream>>>(A, lda, Wt, ldw, bias, Cf, Cb, ldc, K, waves_n, relu); break;
        case 3: gemm_wmma_bf16<3><<<blocks, 256, 0, stream>>>(A, lda, Wt, ldw, bias, Cf, Cb, ldc, K, waves_n, relu); break;
        default: gemm_wmma_bf16<4><<<blocks, 256, 0, stream>>>(A, lda, Wt, ldw, bias, Cf, Cb, ldc, K, waves_n, relu); break;
        }
    };

    // ---- one-time prep ----
    L1D(pack_wcat, NG * KCAT, W_ih, W_hh, wcat);
    L1D(cvt_bf16, H_ * H_, obj_W1, oW1, H_ * H_);
    L1D(cvt_bf16, H_ * H_, del_W1, dW1, H_ * H_);
    L1D(cvt_bf16, H_ * H_, del_W2, dW2, H_ * H_);
    L1D(cvt_bf16, J_ * H_, del_W3, dW3, J_ * H_);
    L1D(cvt_bf16, POSD * H_, var_W, vW, POSD * H_);
    L1D(pack_objW2, 16 * H_, obj_W2, oW2p);
    L1D(cvt_bf16, H_ * Z_, zh_W, zhWb, H_ * Z_);
    L1D(cvt_bf16, H_ * Z_, zc_W, zcWb, H_ * Z_);
    L1D(cvt_bf16, B_ * Z_, z, z_bf, B_ * Z_);
    L1D(vec_add, NG, b_ih, b_hh, gbias, NG);
    L1D(pad_b2f, 16, obj_b2, ob2p);
    L1D(init_prev, B_ * J_, init_a, prev);

    // ---- h0 / c0 ----
    gemm(z_bf, Z_, zhWb, Z_, zh_b, nullptr, h_bf, H_, B_, H_, Z_, 0);
    gemm(z_bf, Z_, zcWb, Z_, zc_b, cst,    nullptr, H_, B_, H_, Z_, 0);

    float* jt_out = out + (size_t)B_ * T_ * POSD;                    // joint_traj segment
    float* ls_out = jt_out + (size_t)B_ * T_ * J_;                   // log-sigma segment

    // ---- recurrent steps ----
    for (int t = 0; t < T_; ++t) {
        L1D(assemble_xh, B_ * KCAT, prev, z, h_bf, jlo, jup, xh, t);
        gemm(xh, KCAT, wcat, KCAT, gbias, gates, nullptr, NG, B_, NG, KCAT, 0);
        L1D(lstm_cell, B_ * H_, gates, cst, h_bf);
        gemm(h_bf, H_, oW1,  H_, obj_b1, nullptr, t1, H_, B_, H_, H_, 1);
        gemm(t1,   H_, oW2p, H_, ob2p,   objl, nullptr, 16, B_, 16, H_, 0);
        gemm(h_bf, H_, dW1,  H_, del_b1, nullptr, t1, H_, B_, H_, H_, 1);
        gemm(t1,   H_, dW2,  H_, del_b2, nullptr, t2, H_, B_, H_, H_, 1);
        gemm(t2,   H_, dW3,  H_, del_b3, rdel, nullptr, J_, B_, J_, H_, 0);
        gemm(h_bf, H_, vW,   H_, var_b,  vlin + t * POSD, nullptr, T_ * POSD, B_, POSD, H_, 0);
        L1D(joint_update, B_ * J_, prev, rdel, dscale, jlo, jup, objl, jt_out, otr, t);
    }

    // ---- FK + normalize + log-sigma ----
    L1D(final_post, B_ * T_, jt_out, otr, vlin, fk_W, fk_b, pmean, pstd, out, ls_out);

    #undef L1D
}
